// HybridEstimatorQNN_65481071407721
// MI455X (gfx1250) — matrix operand order repaired and checked by reading, not compiled, not run on hardware
//
#include <hip/hip_runtime.h>
#include <hip/hip_bf16.h>

typedef __bf16 v16bf __attribute__((ext_vector_type(16)));
typedef float  v8f   __attribute__((ext_vector_type(8)));

union FragU { uint4 u[2]; v16bf v; };

static __device__ inline v16bf load_frag(const __bf16* p0, const __bf16* p1) {
    FragU f;
    f.u[0] = *reinterpret_cast<const uint4*>(p0);
    f.u[1] = *reinterpret_cast<const uint4*>(p1);
    return f.v;
}

// LDS byte offset (wave-relative) of a __shared__ object, for async-to-LDS ops
static __device__ inline unsigned lds_off(const void* p) {
    return (unsigned)(size_t)(const __attribute__((address_space(3))) void*)p;
}

// 64B / 32B async global->LDS copies (GVS mode: uniform SGPR base + per-lane u32 offset).
// Tracked by ASYNCcnt; instruction offset applies to BOTH global and LDS address.
#define ASYNC_CP4(ldsoff, voff, sbase)                                       \
    asm volatile("global_load_async_to_lds_b128 %0, %1, %2 offset:0\n\t"     \
                 "global_load_async_to_lds_b128 %0, %1, %2 offset:16\n\t"    \
                 "global_load_async_to_lds_b128 %0, %1, %2 offset:32\n\t"    \
                 "global_load_async_to_lds_b128 %0, %1, %2 offset:48"        \
                 :: "v"(ldsoff), "v"(voff), "s"(sbase) : "memory")

#define ASYNC_CP2(ldsoff, voff, sbase)                                       \
    asm volatile("global_load_async_to_lds_b128 %0, %1, %2 offset:0\n\t"     \
                 "global_load_async_to_lds_b128 %0, %1, %2 offset:16"        \
                 :: "v"(ldsoff), "v"(voff), "s"(sbase) : "memory")

#define ASYNC_WAIT() asm volatile("s_wait_asynccnt 0x0" ::: "memory")

// ---------------- Kernel 1: W1 -> bf16 transposed [128][4096], W2 -> bf16 transposed [64][128]
__global__ __launch_bounds__(256) void prep_kernel(const float* __restrict__ W1,
                                                   const float* __restrict__ W2,
                                                   __bf16* __restrict__ W1t,
                                                   __bf16* __restrict__ W2t) {
    int idx = blockIdx.x * 256 + threadIdx.x;
    if (idx < 4096 * 128) {
        int n = idx >> 12;
        int k = idx & 4095;
        W1t[(size_t)n * 4096 + k] = (__bf16)W1[(size_t)k * 128 + n];
    } else {
        int j = idx - 4096 * 128;
        if (j < 64 * 128) {
            int n = j >> 7;
            int k = j & 127;
            W2t[n * 128 + k] = (__bf16)W2[k * 64 + n];
        }
    }
}

// ---------------- Kernel 2: GEMM1  h1 = relu(x @ W1 + b1); double-buffered, async B staging
__global__ __launch_bounds__(256) void gemm1_kernel(const float* __restrict__ x,
                                                    const __bf16* __restrict__ W1t,
                                                    const float* __restrict__ b1,
                                                    __bf16* __restrict__ h1) {
    __shared__ __bf16 As[2][32 * 64];    // [m][k] tile, KC=64, double buffered (x converted)
    __shared__ __bf16 Bs[2][128 * 64];   // [n][k] tile, double buffered (bf16 copy)
    const int tid  = threadIdx.x;
    const int wave = tid >> 5;
    const int lane = tid & 31;
    const int l    = lane & 15;
    const int hi   = lane >> 4;
    const int mt   = wave & 1;              // 2 M-tiles of 16 (32 rows)
    const int nb   = (wave >> 1) * 2;       // n-tile pair base (8 tiles / 128 cols)
    const int blockRow = blockIdx.x * 32;

    const int arow = tid >> 3, akk = (tid & 7) * 8;   // A: 8 f32 per thread
    const int bn   = tid >> 1, bkk = (tid & 1) * 32;  // B: 32 bf16 per thread

    const uint64_t sbase  = (uint64_t)(size_t)W1t;
    const unsigned bvoff0 = (unsigned)((bn * 4096 + bkk) * 2);
    const unsigned ldsB0  = lds_off(&Bs[0][bn * 64 + bkk]);
    const unsigned ldsB1  = lds_off(&Bs[1][bn * 64 + bkk]);
    const float* gA = x + (size_t)(blockRow + arow) * 4096 + akk;

    // prologue: stage tile k0=0 into buffer 0
    {
        ASYNC_CP4(ldsB0, bvoff0, sbase);
        float4 a0 = ((const float4*)gA)[0];
        float4 a1 = ((const float4*)gA)[1];
        __bf16* da = &As[0][arow * 64 + akk];
        da[0]=(__bf16)a0.x; da[1]=(__bf16)a0.y; da[2]=(__bf16)a0.z; da[3]=(__bf16)a0.w;
        da[4]=(__bf16)a1.x; da[5]=(__bf16)a1.y; da[6]=(__bf16)a1.z; da[7]=(__bf16)a1.w;
        ASYNC_WAIT();
    }
    __syncthreads();

    const v8f zero = {0.f,0.f,0.f,0.f,0.f,0.f,0.f,0.f};
    v8f acc[2] = {zero, zero};
    const int m = mt * 16 + l;

    for (int k0 = 0; k0 < 4096; k0 += 64) {
        const int  cur  = (k0 >> 6) & 1;
        const bool more = (k0 + 64) < 4096;
        float4 na0 = {0,0,0,0}, na1 = {0,0,0,0};
        if (more) {   // prefetch next tile while computing this one
            ASYNC_CP4(cur ? ldsB0 : ldsB1, bvoff0 + (unsigned)(k0 + 64) * 2, sbase);
            const float4* gn = (const float4*)(gA + k0 + 64);
            na0 = gn[0]; na1 = gn[1];
        }
        #pragma unroll
        for (int h = 0; h < 2; ++h) {   // two K=32 sub-chunks
            v16bf af = load_frag(&As[cur][m * 64 + h * 32 + hi * 16],
                                 &As[cur][m * 64 + h * 32 + hi * 16 + 8]);
            #pragma unroll
            for (int j = 0; j < 2; ++j) {
                int n = (nb + j) * 16 + l;
                v16bf bf = load_frag(&Bs[cur][n * 64 + h * 32 + hi * 16],
                                     &Bs[cur][n * 64 + h * 32 + hi * 16 + 8]);
                acc[j] = __builtin_amdgcn_wmma_f32_16x16x32_bf16(
                    false, af, false, bf, (short)0, acc[j], false, false);
            }
        }
        if (more) {   // drain prefetch into the other buffer
            __bf16* da = &As[cur ^ 1][arow * 64 + akk];
            da[0]=(__bf16)na0.x; da[1]=(__bf16)na0.y; da[2]=(__bf16)na0.z; da[3]=(__bf16)na0.w;
            da[4]=(__bf16)na1.x; da[5]=(__bf16)na1.y; da[6]=(__bf16)na1.z; da[7]=(__bf16)na1.w;
            ASYNC_WAIT();
        }
        __syncthreads();
    }
    // epilogue: +bias, relu, store bf16
    #pragma unroll
    for (int j = 0; j < 2; ++j) {
        const int col  = (nb + j) * 16 + l;
        const float bias = b1[col];
        #pragma unroll
        for (int r = 0; r < 8; ++r) {
            int row = blockRow + mt * 16 + r + hi * 8;
            float v = acc[j][r] + bias;
            v = v > 0.f ? v : 0.f;
            h1[(size_t)row * 128 + col] = (__bf16)v;
        }
    }
}

// ---------------- Kernel 3: GEMM2  h2 = relu(h1 @ W2 + b2); fully async-staged LDS
__global__ __launch_bounds__(256) void gemm2_kernel(const __bf16* __restrict__ h1,
                                                    const __bf16* __restrict__ W2t,
                                                    const float* __restrict__ b2,
                                                    float* __restrict__ h2) {
    __shared__ __bf16 As[32 * 128];   // [m][k], full K=128
    __shared__ __bf16 Bs[64 * 128];   // [n][k]
    const int tid  = threadIdx.x;
    const int wave = tid >> 5;
    const int lane = tid & 31;
    const int l    = lane & 15;
    const int hi   = lane >> 4;
    const int mt   = wave & 1;        // 2 M-tiles
    const int nt   = wave >> 1;       // 4 N-tiles
    const int blockRow = blockIdx.x * 32;

    {   // stage A (h1 tile): 32B per thread, async
        int row = tid >> 3, seg = (tid & 7) * 16;
        unsigned ldsA  = lds_off(&As[row * 128 + seg]);
        unsigned avoff = (unsigned)((((size_t)(blockRow + row)) * 128 + seg) * 2);
        ASYNC_CP2(ldsA, avoff, (uint64_t)(size_t)h1);
    }
    {   // stage B (all of W2t): 64B per thread, async
        int row = tid >> 2, seg = (tid & 3) * 32;
        unsigned ldsBo = lds_off(&Bs[row * 128 + seg]);
        unsigned bvoff = (unsigned)((row * 128 + seg) * 2);
        ASYNC_CP4(ldsBo, bvoff, (uint64_t)(size_t)W2t);
    }
    ASYNC_WAIT();
    __syncthreads();

    const v8f zero = {0.f,0.f,0.f,0.f,0.f,0.f,0.f,0.f};
    v8f acc = zero;
    const int m = mt * 16 + l;
    const int n = nt * 16 + l;
    #pragma unroll
    for (int kc = 0; kc < 4; ++kc) {
        v16bf af = load_frag(&As[m * 128 + kc * 32 + hi * 16],
                             &As[m * 128 + kc * 32 + hi * 16 + 8]);
        v16bf bf = load_frag(&Bs[n * 128 + kc * 32 + hi * 16],
                             &Bs[n * 128 + kc * 32 + hi * 16 + 8]);
        acc = __builtin_amdgcn_wmma_f32_16x16x32_bf16(
            false, af, false, bf, (short)0, acc, false, false);
    }
    const int col  = nt * 16 + l;
    const float bias = b2[col];
    #pragma unroll
    for (int r = 0; r < 8; ++r) {
        int row = blockRow + mt * 16 + r + hi * 8;
        float v = acc[r] + bias;
        h2[(size_t)row * 64 + col] = v > 0.f ? v : 0.f;
    }
}

// ---------------- Kernel 4: latent = h2 @ W3 + b3   -> [8192]
__global__ __launch_bounds__(256) void latent_kernel(const float* __restrict__ h2,
                                                     const float* __restrict__ W3,
                                                     const float* __restrict__ b3,
                                                     float* __restrict__ lat) {
    __shared__ float W3s[64];
    __shared__ float b3s;
    int tid = threadIdx.x;
    if (tid < 64) W3s[tid] = W3[tid];
    if (tid == 0) b3s = b3[0];
    __syncthreads();
    int i = blockIdx.x * 256 + tid;
    const float4* r = (const float4*)(h2 + (size_t)i * 64);
    float s = 0.f;
    #pragma unroll
    for (int q = 0; q < 16; ++q) {
        float4 v = r[q];
        s += v.x * W3s[q*4+0] + v.y * W3s[q*4+1] + v.z * W3s[q*4+2] + v.w * W3s[q*4+3];
    }
    lat[i] = s + b3s;
}

// ---------------- Kernel 5: global min/max of latent (for stable softmax)
__global__ __launch_bounds__(1024) void minmax_kernel(const float* __restrict__ lat,
                                                      float* __restrict__ mnx) {
    __shared__ float smin[1024];
    __shared__ float smax[1024];
    int tid = threadIdx.x;
    float mn = 3.4e38f, mx = -3.4e38f;
    #pragma unroll
    for (int r = 0; r < 8; ++r) {
        float v = lat[tid + r * 1024];
        mn = fminf(mn, v);
        mx = fmaxf(mx, v);
    }
    smin[tid] = mn; smax[tid] = mx;
    __syncthreads();
    for (int s = 512; s > 0; s >>= 1) {
        if (tid < s) {
            smin[tid] = fminf(smin[tid], smin[tid + s]);
            smax[tid] = fmaxf(smax[tid], smax[tid + s]);
        }
        __syncthreads();
    }
    if (tid == 0) { mnx[0] = smin[0]; mnx[1] = smax[0]; }
}

// ---------------- Kernel 6: attention partial sums over j-chunks of 1024
__global__ __launch_bounds__(256) void attn_part_kernel(const float* __restrict__ lat,
                                                        const float* __restrict__ Rp,
                                                        const float* __restrict__ Ep,
                                                        const float* __restrict__ mnx,
                                                        float* __restrict__ nump,
                                                        float* __restrict__ denp) {
    __shared__ float sl[1024];
    int tid = threadIdx.x;
    int jc  = blockIdx.y;
    ((float4*)sl)[tid] = ((const float4*)(lat + jc * 1024))[tid];
    __syncthreads();
    int i = blockIdx.x * 256 + tid;
    float li = lat[i];
    float a  = Rp[0] * Ep[0] * li;                 // q_i*k_j = a * lat_j
    float m  = fmaxf(a * mnx[0], a * mnx[1]);      // exact row max over all j
    float num = 0.f, den = 0.f;
    for (int j = 0; j < 1024; ++j) {
        float lj = sl[j];
        float e  = __expf(a * lj - m);
        num = fmaf(e, lj, num);
        den += e;
    }
    nump[jc * 8192 + i] = num;
    denp[jc * 8192 + i] = den;
}

// ---------------- Kernel 7: finalize attention + QFC MLP -> f [8192,4]
__global__ __launch_bounds__(256) void qfc_kernel(const float* __restrict__ nump,
                                                  const float* __restrict__ denp,
                                                  const float* __restrict__ Wc1, const float* __restrict__ bc1,
                                                  const float* __restrict__ Wc2, const float* __restrict__ bc2,
                                                  const float* __restrict__ Wf1, const float* __restrict__ bf1,
                                                  const float* __restrict__ Wf2, const float* __restrict__ bf2,
                                                  float* __restrict__ fb) {
    __shared__ float sWc1[8], sbc1[8], sWc2[128], sbc2[16];
    __shared__ float sWf1[1024], sbf1[64], sWf2[256], sbf2[4];
    int tid = threadIdx.x;
    for (int t = tid; t < 1024; t += 256) sWf1[t] = Wf1[t];
    if (tid < 256) sWf2[tid] = Wf2[tid];
    if (tid < 128) sWc2[tid] = Wc2[tid];
    if (tid < 64)  sbf1[tid] = bf1[tid];
    if (tid < 16)  sbc2[tid] = bc2[tid];
    if (tid < 8) { sWc1[tid] = Wc1[tid]; sbc1[tid] = bc1[tid]; }
    if (tid < 4)   sbf2[tid] = bf2[tid];
    __syncthreads();

    int i = blockIdx.x * 256 + tid;
    float num = 0.f, den = 0.f;
    #pragma unroll
    for (int jc = 0; jc < 8; ++jc) { num += nump[jc * 8192 + i]; den += denp[jc * 8192 + i]; }
    float attn = num / den;

    float c1[8], c2[16], fh[64];
    #pragma unroll
    for (int o = 0; o < 8; ++o) {
        float s = attn * sWc1[o] + sbc1[o];
        c1[o] = s > 0.f ? s : 0.f;
    }
    #pragma unroll
    for (int o = 0; o < 16; ++o) {
        float s = sbc2[o];
        #pragma unroll
        for (int u = 0; u < 8; ++u) s = fmaf(c1[u], sWc2[u * 16 + o], s);
        c2[o] = s > 0.f ? s : 0.f;
    }
    for (int o = 0; o < 64; ++o) {
        float s = sbf1[o];
        #pragma unroll
        for (int u = 0; u < 16; ++u) s = fmaf(c2[u], sWf1[u * 64 + o], s);
        fh[o] = s > 0.f ? s : 0.f;
    }
    float4 out4;
    float* op = (float*)&out4;
    #pragma unroll
    for (int o = 0; o < 4; ++o) {
        float s = sbf2[o];
        for (int u = 0; u < 64; ++u) s = fmaf(fh[u], sWf2[u * 4 + o], s);
        op[o] = s;
    }
    ((float4*)fb)[i] = out4;
}

// ---------------- Kernel 8: batch statistics (mean / rstd per column)
__global__ __launch_bounds__(1024) void stats_kernel(const float* __restrict__ fb,
                                                     float* __restrict__ stat) {
    __shared__ float red[8 * 1024];   // 32 KB
    int tid = threadIdx.x;
    float s[4]  = {0,0,0,0};
    float sq[4] = {0,0,0,0};
    #pragma unroll
    for (int r = 0; r < 8; ++r) {
        float4 v = ((const float4*)fb)[tid + r * 1024];
        s[0] += v.x; sq[0] += v.x * v.x;
        s[1] += v.y; sq[1] += v.y * v.y;
        s[2] += v.z; sq[2] += v.z * v.z;
        s[3] += v.w; sq[3] += v.w * v.w;
    }
    #pragma unroll
    for (int c = 0; c < 4; ++c) { red[c * 1024 + tid] = s[c]; red[(4 + c) * 1024 + tid] = sq[c]; }
    __syncthreads();
    for (int st = 512; st > 0; st >>= 1) {
        if (tid < st) {
            #pragma unroll
            for (int c = 0; c < 8; ++c) red[c * 1024 + tid] += red[c * 1024 + tid + st];
        }
        __syncthreads();
    }
    if (tid < 4) {
        float mu  = red[tid * 1024] * (1.f / 8192.f);
        float var = red[(4 + tid) * 1024] * (1.f / 8192.f) - mu * mu;
        stat[tid]     = mu;
        stat[4 + tid] = rsqrtf(var + 1e-5f);
    }
}

// ---------------- Kernel 9: batchnorm + tanh regressor head -> out [8192]
__global__ __launch_bounds__(256) void head_kernel(const float* __restrict__ fb,
                                                   const float* __restrict__ stat,
                                                   const float* __restrict__ gamma,
                                                   const float* __restrict__ beta,
                                                   const float* __restrict__ Wr1,
                                                   const float* __restrict__ br1,
                                                   const float* __restrict__ Wr2,
                                                   const float* __restrict__ br2,
                                                   float* __restrict__ out) {
    int i = blockIdx.x * 256 + threadIdx.x;
    float4 fv = ((const float4*)fb)[i];
    float f[4] = {fv.x, fv.y, fv.z, fv.w};
    float fn[4];
    #pragma unroll
    for (int c = 0; c < 4; ++c)
        fn[c] = (f[c] - stat[c]) * stat[4 + c] * gamma[c] + beta[c];
    float o = br2[0];
    #pragma unroll
    for (int k = 0; k < 8; ++k) {
        float z = br1[k];
        #pragma unroll
        for (int c = 0; c < 4; ++c) z = fmaf(fn[c], Wr1[c * 8 + k], z);
        o = fmaf(tanhf(z), Wr2[k], o);
    }
    out[i] = o;
}

extern "C" void kernel_launch(void* const* d_in, const int* in_sizes, int n_in,
                              void* d_out, int out_size, void* d_ws, size_t ws_size,
                              hipStream_t stream) {
    (void)in_sizes; (void)n_in; (void)out_size; (void)ws_size;
    const float* x     = (const float*)d_in[0];
    const float* W1    = (const float*)d_in[1];
    const float* b1    = (const float*)d_in[2];
    const float* W2    = (const float*)d_in[3];
    const float* b2    = (const float*)d_in[4];
    const float* W3    = (const float*)d_in[5];
    const float* b3    = (const float*)d_in[6];
    const float* R     = (const float*)d_in[7];
    const float* E     = (const float*)d_in[8];
    const float* Wc1   = (const float*)d_in[9];
    const float* bc1   = (const float*)d_in[10];
    const float* Wc2   = (const float*)d_in[11];
    const float* bc2   = (const float*)d_in[12];
    const float* Wf1   = (const float*)d_in[13];
    const float* bf1   = (const float*)d_in[14];
    const float* Wf2   = (const float*)d_in[15];
    const float* bf2   = (const float*)d_in[16];
    const float* gamma = (const float*)d_in[17];
    const float* beta  = (const float*)d_in[18];
    const float* Wr1   = (const float*)d_in[19];
    const float* br1   = (const float*)d_in[20];
    const float* Wr2   = (const float*)d_in[21];
    const float* br2   = (const float*)d_in[22];
    float* out = (float*)d_out;

    char* w = (char*)d_ws;
    __bf16* W1T = (__bf16*)(w + 0x000000);   // 1 MB
    __bf16* W2T = (__bf16*)(w + 0x100000);   // 16 KB
    __bf16* H1  = (__bf16*)(w + 0x110000);   // 2 MB
    float*  H2  = (float*) (w + 0x310000);   // 2 MB
    float*  LAT = (float*) (w + 0x510000);   // 32 KB
    float*  MNX = (float*) (w + 0x518000);   // 256 B
    float*  NUP = (float*) (w + 0x518100);   // 256 KB
    float*  DEP = (float*) (w + 0x558100);   // 256 KB
    float*  FB  = (float*) (w + 0x598100);   // 128 KB
    float*  ST  = (float*) (w + 0x5B8100);   // 64 B

    prep_kernel<<<2080, 256, 0, stream>>>(W1, W2, W1T, W2T);
    gemm1_kernel<<<256, 256, 0, stream>>>(x, W1T, b1, H1);
    gemm2_kernel<<<256, 256, 0, stream>>>(H1, W2T, b2, H2);
    latent_kernel<<<32, 256, 0, stream>>>(H2, W3, b3, LAT);
    minmax_kernel<<<1, 1024, 0, stream>>>(LAT, MNX);
    attn_part_kernel<<<dim3(32, 8), 256, 0, stream>>>(LAT, R, E, MNX, NUP, DEP);
    qfc_kernel<<<32, 256, 0, stream>>>(NUP, DEP, Wc1, bc1, Wc2, bc2, Wf1, bf1, Wf2, bf2, FB);
    stats_kernel<<<1, 1024, 0, stream>>>(FB, ST);
    head_kernel<<<32, 256, 0, stream>>>(FB, ST, gamma, beta, Wr1, br1, Wr2, br2, out);
}
